// _PTSparse4DHead_22411139350622
// MI455X (gfx1250) — matrix operand order repaired and checked by reading, not compiled, not run on hardware
//
#include <hip/hip_runtime.h>

// ---------------- problem constants ----------------
#define NA   900
#define NCAM 6
#define NL   4
#define NPTS 13
#define ED   256
#define NG   8
#define GD   32
#define NOUT 416              // NG*NL*NPTS
#define MROWS (NA*NCAM)       // 5400
#define SM_N  (NCAM*NL*NPTS)  // 312 softmax extent

typedef __attribute__((ext_vector_type(16))) _Float16 v16h;
typedef __attribute__((ext_vector_type(2)))  _Float16 h2;
typedef __attribute__((ext_vector_type(8)))  float    v8f;

// ---------------- workspace layout (floats) ----------------
#define CAM_OFF    0            // 6*256            = 1536
#define KP_OFF     4096         // 900*13*3         = 35100
#define GRID_OFF   40960        // 6*900*13*2       = 140400
#define LOGITS_OFF 200704       // 5400*416         = 2246400
#define FUSED_OFF  2447104      // 900*256          = 230400
#define TR_OFF     2709248      // transposed fmaps (optional, ws-gated)
#define TR0_SZ     (6*ED*64*176)   // 17301504
#define TR1_SZ     (6*ED*32*88)    //  4325376
#define TR2_SZ     (6*ED*16*44)    //  1081344
#define TR3_SZ     (6*ED*8*22)     //   270336
#define TR_TOTAL   (TR0_SZ+TR1_SZ+TR2_SZ+TR3_SZ)   // 22978560

// ---------------- helpers ----------------
__device__ __forceinline__ float block_sum(float x, float* red, int t, int n) {
  red[t] = x; __syncthreads();
  for (int s = n >> 1; s > 0; s >>= 1) { if (t < s) red[t] += red[t + s]; __syncthreads(); }
  float r = red[0]; __syncthreads();
  return r;
}
__device__ __forceinline__ float block_max(float x, float* red, int t, int n) {
  red[t] = x; __syncthreads();
  for (int s = n >> 1; s > 0; s >>= 1) { if (t < s) red[t] = fmaxf(red[t], red[t + s]); __syncthreads(); }
  float r = red[0]; __syncthreads();
  return r;
}

// A-fragment per ISA 7.12.2: lane holds row (lane&15), K pairs at
// kb + {0,2,4,6} and kb+16+{0,2,4,6}, kb = (lane>>4)*8.  LDS row-major [m][32].
__device__ __forceinline__ v16h load_afrag(const _Float16* ldsA, int lane) {
  v16h a;
  int m = lane & 15, kb = (lane >> 4) * 8;
#pragma unroll
  for (int q = 0; q < 8; ++q) {
    int k = kb + ((q < 4) ? (2 * q) : (16 + 2 * (q - 4)));
    h2 p = *(const h2*)&ldsA[m * 32 + k];
    a[2 * q] = p[0]; a[2 * q + 1] = p[1];
  }
  return a;
}
// B-fragment: lane holds col n, K pairs at kb + {0,2,4,6} and kb+16+{0,2,4,6}, kb = (lane>>4)*8.  LDS col-major [n][32].
__device__ __forceinline__ v16h load_bfrag(const _Float16* ldsB, int lane, int nloc) {
  v16h b;
  int n = nloc + (lane & 15), kb = (lane >> 4) * 8;
#pragma unroll
  for (int q = 0; q < 8; ++q) {
    h2 p = *(const h2*)&ldsB[n * 32 + kb + ((q < 4) ? (2 * q) : (16 + 2 * (q - 4)))];
    b[2 * q] = p[0]; b[2 * q + 1] = p[1];
  }
  return b;
}

// ================= kernel 1: camera embed MLP (6 blocks x 256) =================
__global__ __launch_bounds__(256) void cam_embed_kernel(
    const float* __restrict__ proj, const float* __restrict__ w1, const float* __restrict__ b1,
    const float* __restrict__ g1, const float* __restrict__ be1,
    const float* __restrict__ w2, const float* __restrict__ b2,
    const float* __restrict__ g2, const float* __restrict__ be2,
    float* __restrict__ cam_embed) {
  __shared__ float sh[ED];
  __shared__ float red[ED];
  int c = blockIdx.x, t = threadIdx.x;
  float cin[12];
#pragma unroll
  for (int k = 0; k < 12; ++k) cin[k] = proj[c * 16 + k];
  float h = b1[t];
#pragma unroll
  for (int k = 0; k < 12; ++k) h += cin[k] * w1[k * ED + t];
  h = fmaxf(h, 0.0f);
  float m = block_sum(h, red, t, ED) * (1.0f / ED);
  float d = h - m;
  float v = block_sum(d * d, red, t, ED) * (1.0f / ED);
  h = d * rsqrtf(v + 1e-5f) * g1[t] + be1[t];
  sh[t] = h; __syncthreads();
  float h2v = b2[t];
  for (int k = 0; k < ED; ++k) h2v += sh[k] * w2[k * ED + t];
  h2v = fmaxf(h2v, 0.0f);
  m = block_sum(h2v, red, t, ED) * (1.0f / ED);
  d = h2v - m;
  v = block_sum(d * d, red, t, ED) * (1.0f / ED);
  cam_embed[c * ED + t] = d * rsqrtf(v + 1e-5f) * g2[t] + be2[t];
}

// ================= kernel 2: lfc sigmoid + key points (900 blocks x 32) =================
__global__ __launch_bounds__(32) void keypoints_kernel(
    const float* __restrict__ inst, const float* __restrict__ anchor,
    const float* __restrict__ fix_scale, const float* __restrict__ lfc_w,
    const float* __restrict__ lfc_b, float* __restrict__ kp_out) {
  __shared__ float ls[18];
  int a = blockIdx.x, t = threadIdx.x;
  if (t < 18) {
    float s = lfc_b[t];
    const float* f = inst + a * ED;
    for (int k = 0; k < ED; ++k) s += f[k] * lfc_w[k * 18 + t];
    ls[t] = 1.0f / (1.0f + expf(-s)) - 0.5f;
  }
  __syncthreads();
  if (t < NPTS) {
    float sx = expf(anchor[a * 11 + 3]);
    float sy = expf(anchor[a * 11 + 4]);
    float sz = expf(anchor[a * 11 + 5]);
    float kx, ky, kz;
    if (t < 7) { kx = fix_scale[t*3+0]*sx; ky = fix_scale[t*3+1]*sy; kz = fix_scale[t*3+2]*sz; }
    else       { int j = t - 7; kx = ls[j*3+0]*sx; ky = ls[j*3+1]*sy; kz = ls[j*3+2]*sz; }
    float sn = anchor[a * 11 + 6], cs = anchor[a * 11 + 7];
    kp_out[(a * NPTS + t) * 3 + 0] = cs * kx - sn * ky + anchor[a * 11 + 0];
    kp_out[(a * NPTS + t) * 3 + 1] = sn * kx + cs * ky + anchor[a * 11 + 1];
    kp_out[(a * NPTS + t) * 3 + 2] = kz + anchor[a * 11 + 2];
  }
}

// ================= kernel 3: project to normalized grid coords =================
__global__ __launch_bounds__(256) void project_kernel(
    const float* __restrict__ kp, const float* __restrict__ proj,
    const float* __restrict__ image_wh, float* __restrict__ gridbuf) {
  int idx = blockIdx.x * blockDim.x + threadIdx.x;
  if (idx >= NCAM * NA * NPTS) return;
  int p = idx % NPTS;
  int a = (idx / NPTS) % NA;
  int c = idx / (NA * NPTS);
  float x = kp[(a * NPTS + p) * 3 + 0];
  float y = kp[(a * NPTS + p) * 3 + 1];
  float z = kp[(a * NPTS + p) * 3 + 2];
  const float* P = proj + c * 16;
  float px = P[0]*x + P[1]*y + P[2]*z  + P[3];
  float py = P[4]*x + P[5]*y + P[6]*z  + P[7];
  float pz = P[8]*x + P[9]*y + P[10]*z + P[11];
  float dd = fmaxf(pz, 1e-5f);
  float u = px / dd / image_wh[c * 2 + 0];
  float v = py / dd / image_wh[c * 2 + 1];
  gridbuf[idx * 2 + 0] = u * 2.0f - 1.0f;
  gridbuf[idx * 2 + 1] = v * 2.0f - 1.0f;
}

// ================= WMMA GEMM: wts logits, 64x64 block tile, 4 WMMA/wave/K-step ==========
__global__ __launch_bounds__(128) void wts_gemm_kernel(
    const float* __restrict__ inst, const float* __restrict__ aemb,
    const float* __restrict__ cam_embed, const float* __restrict__ wfc_w,
    const float* __restrict__ wfc_b, float* __restrict__ logits) {
  __shared__ _Float16 ldsA[64 * 32];   // row-major [m][k]
  __shared__ _Float16 ldsB[64 * 32];   // col-major [n][k]
  int tid = threadIdx.x, wave = tid >> 5, lane = tid & 31;
  int m0 = blockIdx.x * 64;
  int n0 = blockIdx.y * 64;
  v8f acc0 = {}, acc1 = {}, acc2 = {}, acc3 = {};
  for (int ks = 0; ks < ED; ks += 32) {
    for (int i = tid; i < 2048; i += 128) {          // A: 64 rows x 32 k
      int r = i >> 5, k = i & 31;
      int row = m0 + r;
      float v = 0.0f;
      if (row < MROWS) {
        int a = row / NCAM, c = row % NCAM;
        v = inst[a * ED + ks + k] + aemb[a * ED + ks + k] + cam_embed[c * ED + ks + k];
      }
      ldsA[r * 32 + k] = (_Float16)v;
    }
    for (int i = tid; i < 2048; i += 128) {          // B: 32 k x 64 n (stored [n][k])
      int nloc = i >> 5, k = i & 31;
      int col = n0 + nloc;
      float v = (col < NOUT) ? wfc_w[(ks + k) * NOUT + col] : 0.0f;
      ldsB[nloc * 32 + k] = (_Float16)v;
    }
    if (ks + 32 < ED) {  // prefetch next B K-tile -> global_prefetch_b8
      int col = n0 + (tid & 63);
      if (col < NOUT) __builtin_prefetch(&wfc_w[(ks + 32 + (tid >> 6) * 16) * NOUT + col], 0, 1);
    }
    __syncthreads();
    v16h afrag = load_afrag(&ldsA[(wave << 4) * 32], lane);   // wave's 16-row M slice
    v16h b0 = load_bfrag(ldsB, lane, 0);
    v16h b1 = load_bfrag(ldsB, lane, 16);
    v16h b2 = load_bfrag(ldsB, lane, 32);
    v16h b3 = load_bfrag(ldsB, lane, 48);
    acc0 = __builtin_amdgcn_wmma_f32_16x16x32_f16(false, afrag, false, b0, (short)0, acc0, false, false);
    acc1 = __builtin_amdgcn_wmma_f32_16x16x32_f16(false, afrag, false, b1, (short)0, acc1, false, false);
    acc2 = __builtin_amdgcn_wmma_f32_16x16x32_f16(false, afrag, false, b2, (short)0, acc2, false, false);
    acc3 = __builtin_amdgcn_wmma_f32_16x16x32_f16(false, afrag, false, b3, (short)0, acc3, false, false);
    __syncthreads();
  }
  int mbase = m0 + (wave << 4) + (lane >> 4) * 8;
  v8f accs[4] = {acc0, acc1, acc2, acc3};
#pragma unroll
  for (int j = 0; j < 4; ++j) {
    int n = n0 + j * 16 + (lane & 15);
    if (n < NOUT) {
      float bb = wfc_b[n];
#pragma unroll
      for (int r = 0; r < 8; ++r) {
        int m = mbase + r;
        if (m < MROWS) logits[m * NOUT + n] = accs[j][r] + bb;
      }
    }
  }
}

// ================= kernel 5: softmax over 312 per (anchor, group), in place =================
__global__ __launch_bounds__(64) void softmax_kernel(float* __restrict__ logits) {
  __shared__ float red[64];
  int a = blockIdx.x, g = blockIdx.y, t = threadIdx.x;
  float mx = -3.4e38f;
  for (int i = t; i < SM_N; i += 64) {
    int c = i / (NL * NPTS), r = i % (NL * NPTS);
    mx = fmaxf(mx, logits[(a * NCAM + c) * NOUT + r * NG + g]);
  }
  mx = block_max(mx, red, t, 64);
  float sum = 0.0f;
  for (int i = t; i < SM_N; i += 64) {
    int c = i / (NL * NPTS), r = i % (NL * NPTS);
    int off = (a * NCAM + c) * NOUT + r * NG + g;
    float e = expf(logits[off] - mx);
    logits[off] = e;
    sum += e;
  }
  sum = block_sum(sum, red, t, 64);
  float inv = 1.0f / sum;
  for (int i = t; i < SM_N; i += 64) {
    int c = i / (NL * NPTS), r = i % (NL * NPTS);
    logits[(a * NCAM + c) * NOUT + r * NG + g] *= inv;
  }
}

// ============ kernel T: tiled transpose (ED, HW) -> (HW, ED) per camera ============
__global__ __launch_bounds__(256) void transpose_kernel(
    const float* __restrict__ src, float* __restrict__ dst, int HW) {
  __shared__ float tile[32][33];
  int c = blockIdx.z;
  int p0 = blockIdx.x * 32;
  int ch0 = blockIdx.y * 32;
  int tx = threadIdx.x, ty = threadIdx.y;   // (32, 8)
  for (int i = ty; i < 32; i += 8) {        // coalesced along pixels
    int p = p0 + tx;
    tile[i][tx] = (p < HW) ? src[((size_t)c * ED + ch0 + i) * HW + p] : 0.0f;
  }
  __syncthreads();
  for (int i = ty; i < 32; i += 8) {        // coalesced along channels
    int p = p0 + i;
    if (p < HW) dst[((size_t)c * HW + p) * ED + ch0 + tx] = tile[tx][i];
  }
}

// ===== kernel 6a: fuse from TRANSPOSED fmaps (c,H,W,ED): coalesced corner loads =====
__global__ __launch_bounds__(256) void fuse_tr_kernel(
    const float* __restrict__ tr0, const float* __restrict__ tr1,
    const float* __restrict__ tr2, const float* __restrict__ tr3,
    const float* __restrict__ gridbuf, const float* __restrict__ wts,
    float* __restrict__ fused) {
  int a = blockIdx.x, ch = threadIdx.x;
  int g = ch >> 5;
  const float* trs[4] = {tr0, tr1, tr2, tr3};
  const int Hs[4] = {64, 32, 16, 8};
  const int Ws[4] = {176, 88, 44, 22};
  float acc = 0.0f;
  for (int c = 0; c < NCAM; ++c) {
    for (int p = 0; p < NPTS; ++p) {
      int gi = ((c * NA + a) * NPTS + p) * 2;
      float gxn = gridbuf[gi + 0];
      float gyn = gridbuf[gi + 1];
#pragma unroll
      for (int l = 0; l < NL; ++l) {
        const int H = Hs[l], W = Ws[l];
        float gx = fminf(fmaxf((gxn + 1.0f) * (W * 0.5f) - 0.5f, -10000.0f), 10000.0f);
        float gy = fminf(fmaxf((gyn + 1.0f) * (H * 0.5f) - 0.5f, -10000.0f), 10000.0f);
        float x0 = floorf(gx), y0 = floorf(gy);
        float wx1 = gx - x0, wy1 = gy - y0;
        float wx0 = 1.0f - wx1, wy0 = 1.0f - wy1;
        int x0i = (int)x0, y0i = (int)y0;
        const float* base = trs[l] + (size_t)c * H * W * ED + ch;
        float s = 0.0f;
#pragma unroll
        for (int cy = 0; cy < 2; ++cy) {
#pragma unroll
          for (int cx = 0; cx < 2; ++cx) {
            int xi = x0i + cx, yi = y0i + cy;
            bool valid = (xi >= 0) & (xi < W) & (yi >= 0) & (yi < H);
            int xc = min(max(xi, 0), W - 1), yc = min(max(yi, 0), H - 1);
            float wgt = (cx ? wx1 : wx0) * (cy ? wy1 : wy0);
            float val = base[((size_t)yc * W + xc) * ED];
            s += val * (valid ? wgt : 0.0f);
          }
        }
        acc += s * wts[(a * NCAM + c) * NOUT + (l * NPTS + p) * NG + g];
      }
    }
  }
  fused[a * ED + ch] = acc;
}

// ===== kernel 6b: fallback fuse from original channel-major fmaps =====
__global__ __launch_bounds__(256) void fuse_kernel(
    const float* __restrict__ fmap0, const float* __restrict__ fmap1,
    const float* __restrict__ fmap2, const float* __restrict__ fmap3,
    const float* __restrict__ gridbuf, const float* __restrict__ wts,
    float* __restrict__ fused) {
  int a = blockIdx.x, ch = threadIdx.x;
  int g = ch >> 5;
  const float* fms[4] = {fmap0, fmap1, fmap2, fmap3};
  const int Hs[4] = {64, 32, 16, 8};
  const int Ws[4] = {176, 88, 44, 22};
  float acc = 0.0f;
  for (int c = 0; c < NCAM; ++c) {
    for (int p = 0; p < NPTS; ++p) {
      int gi = ((c * NA + a) * NPTS + p) * 2;
      float gxn = gridbuf[gi + 0];
      float gyn = gridbuf[gi + 1];
#pragma unroll
      for (int l = 0; l < NL; ++l) {
        const int H = Hs[l], W = Ws[l];
        float gx = fminf(fmaxf((gxn + 1.0f) * (W * 0.5f) - 0.5f, -10000.0f), 10000.0f);
        float gy = fminf(fmaxf((gyn + 1.0f) * (H * 0.5f) - 0.5f, -10000.0f), 10000.0f);
        float x0 = floorf(gx), y0 = floorf(gy);
        float wx1 = gx - x0, wy1 = gy - y0;
        float wx0 = 1.0f - wx1, wy0 = 1.0f - wy1;
        int x0i = (int)x0, y0i = (int)y0;
        const float* base = fms[l] + ((size_t)(c * ED + ch)) * H * W;
        float s = 0.0f;
#pragma unroll
        for (int cy = 0; cy < 2; ++cy) {
#pragma unroll
          for (int cx = 0; cx < 2; ++cx) {
            int xi = x0i + cx, yi = y0i + cy;
            bool valid = (xi >= 0) & (xi < W) & (yi >= 0) & (yi < H);
            int xc = min(max(xi, 0), W - 1), yc = min(max(yi, 0), H - 1);
            float wgt = (cx ? wx1 : wx0) * (cy ? wy1 : wy0);
            float val = base[yc * W + xc];
            s += val * (valid ? wgt : 0.0f);
          }
        }
        acc += s * wts[(a * NCAM + c) * NOUT + (l * NPTS + p) * NG + g];
      }
    }
  }
  fused[a * ED + ch] = acc;
}

// ======== WMMA GEMM: out = fused @ op_w + op_b ; fused concat store ========
__global__ __launch_bounds__(128) void out_gemm_kernel(
    const float* __restrict__ fused, const float* __restrict__ op_w,
    const float* __restrict__ op_b, const float* __restrict__ inst,
    float* __restrict__ out) {
  __shared__ _Float16 ldsA[64 * 32];
  __shared__ _Float16 ldsB[64 * 32];
  int tid = threadIdx.x, wave = tid >> 5, lane = tid & 31;
  int m0 = blockIdx.x * 64;
  int n0 = blockIdx.y * 64;
  v8f acc0 = {}, acc1 = {}, acc2 = {}, acc3 = {};
  for (int ks = 0; ks < ED; ks += 32) {
    for (int i = tid; i < 2048; i += 128) {
      int r = i >> 5, k = i & 31;
      int row = m0 + r;
      ldsA[r * 32 + k] = (_Float16)((row < NA) ? fused[row * ED + ks + k] : 0.0f);
    }
    for (int i = tid; i < 2048; i += 128) {
      int nloc = i >> 5, k = i & 31;
      ldsB[nloc * 32 + k] = (_Float16)op_w[(ks + k) * ED + n0 + nloc];
    }
    if (ks + 32 < ED)
      __builtin_prefetch(&op_w[(ks + 32 + (tid >> 6) * 16) * ED + n0 + (tid & 63)], 0, 1);
    __syncthreads();
    v16h afrag = load_afrag(&ldsA[(wave << 4) * 32], lane);
    v16h b0 = load_bfrag(ldsB, lane, 0);
    v16h b1 = load_bfrag(ldsB, lane, 16);
    v16h b2 = load_bfrag(ldsB, lane, 32);
    v16h b3 = load_bfrag(ldsB, lane, 48);
    acc0 = __builtin_amdgcn_wmma_f32_16x16x32_f16(false, afrag, false, b0, (short)0, acc0, false, false);
    acc1 = __builtin_amdgcn_wmma_f32_16x16x32_f16(false, afrag, false, b1, (short)0, acc1, false, false);
    acc2 = __builtin_amdgcn_wmma_f32_16x16x32_f16(false, afrag, false, b2, (short)0, acc2, false, false);
    acc3 = __builtin_amdgcn_wmma_f32_16x16x32_f16(false, afrag, false, b3, (short)0, acc3, false, false);
    __syncthreads();
  }
  int mbase = m0 + (wave << 4) + (lane >> 4) * 8;
  v8f accs[4] = {acc0, acc1, acc2, acc3};
#pragma unroll
  for (int j = 0; j < 4; ++j) {
    int n = n0 + j * 16 + (lane & 15);
    float bb = op_b[n];
#pragma unroll
    for (int r = 0; r < 8; ++r) {
      int m = mbase + r;
      if (m < NA) {
        out[m * (2 * ED) + n] = accs[j][r] + bb;          // out half
        out[m * (2 * ED) + ED + n] = inst[m * ED + n];    // instance_feature half
      }
    }
  }
}

// ================= launch =================
extern "C" void kernel_launch(void* const* d_in, const int* in_sizes, int n_in,
                              void* d_out, int out_size, void* d_ws, size_t ws_size,
                              hipStream_t stream) {
  const float* inst    = (const float*)d_in[0];
  const float* anchor  = (const float*)d_in[1];
  const float* aemb    = (const float*)d_in[2];
  const float* fmap0   = (const float*)d_in[3];
  const float* fmap1   = (const float*)d_in[4];
  const float* fmap2   = (const float*)d_in[5];
  const float* fmap3   = (const float*)d_in[6];
  const float* projm   = (const float*)d_in[7];
  const float* imwh    = (const float*)d_in[8];
  const float* fixsc   = (const float*)d_in[9];
  const float* lfc_w   = (const float*)d_in[10];
  const float* lfc_b   = (const float*)d_in[11];
  const float* cam_w1  = (const float*)d_in[12];
  const float* cam_b1  = (const float*)d_in[13];
  const float* cam_g1  = (const float*)d_in[14];
  const float* cam_be1 = (const float*)d_in[15];
  const float* cam_w2  = (const float*)d_in[16];
  const float* cam_b2  = (const float*)d_in[17];
  const float* cam_g2  = (const float*)d_in[18];
  const float* cam_be2 = (const float*)d_in[19];
  const float* wfc_w   = (const float*)d_in[20];
  const float* wfc_b   = (const float*)d_in[21];
  const float* op_w    = (const float*)d_in[22];
  const float* op_b    = (const float*)d_in[23];
  float* out = (float*)d_out;
  float* ws  = (float*)d_ws;

  float* cam_embed = ws + CAM_OFF;
  float* kp        = ws + KP_OFF;
  float* gridbuf   = ws + GRID_OFF;
  float* logits    = ws + LOGITS_OFF;
  float* fusedb    = ws + FUSED_OFF;

  cam_embed_kernel<<<NCAM, 256, 0, stream>>>(projm, cam_w1, cam_b1, cam_g1, cam_be1,
                                             cam_w2, cam_b2, cam_g2, cam_be2, cam_embed);
  keypoints_kernel<<<NA, 32, 0, stream>>>(inst, anchor, fixsc, lfc_w, lfc_b, kp);
  project_kernel<<<(NCAM * NA * NPTS + 255) / 256, 256, 0, stream>>>(kp, projm, imwh, gridbuf);
  wts_gemm_kernel<<<dim3((MROWS + 63) / 64, (NOUT + 63) / 64), 128, 0, stream>>>(
      inst, aemb, cam_embed, wfc_w, wfc_b, logits);
  softmax_kernel<<<dim3(NA, NG), 64, 0, stream>>>(logits);

  // L2-friendly path: transpose fmaps to (c,H,W,ED) so corner gathers coalesce.
  size_t need = (size_t)(TR_OFF + TR_TOTAL) * sizeof(float);
  if (ws_size >= need) {
    float* tr0 = ws + TR_OFF;
    float* tr1 = tr0 + TR0_SZ;
    float* tr2 = tr1 + TR1_SZ;
    float* tr3 = tr2 + TR2_SZ;
    dim3 blk(32, 8);
    transpose_kernel<<<dim3((64*176 + 31) / 32, ED / 32, NCAM), blk, 0, stream>>>(fmap0, tr0, 64*176);
    transpose_kernel<<<dim3((32*88  + 31) / 32, ED / 32, NCAM), blk, 0, stream>>>(fmap1, tr1, 32*88);
    transpose_kernel<<<dim3((16*44  + 31) / 32, ED / 32, NCAM), blk, 0, stream>>>(fmap2, tr2, 16*44);
    transpose_kernel<<<dim3((8*22   + 31) / 32, ED / 32, NCAM), blk, 0, stream>>>(fmap3, tr3, 8*22);
    fuse_tr_kernel<<<NA, 256, 0, stream>>>(tr0, tr1, tr2, tr3, gridbuf, logits, fusedb);
  } else {
    fuse_kernel<<<NA, 256, 0, stream>>>(fmap0, fmap1, fmap2, fmap3, gridbuf, logits, fusedb);
  }

  out_gemm_kernel<<<dim3((NA + 63) / 64, ED / 64), 128, 0, stream>>>(
      fusedb, op_w, op_b, inst, out);
}
